// Decoder_48670569398470
// MI455X (gfx1250) — matrix operand order repaired and checked by reading, not compiled, not run on hardware
//
#include <hip/hip_runtime.h>
#include <hip/hip_bf16.h>

// ---------------------------------------------------------------------------
// Tacotron2-style decoder for MI455X (gfx1250, wave32, WMMA bf16->f32).
// bf16 operands for all GEMMs (recurrent weights ~36.5MB bf16 -> L2-resident
// across the 800-step scan on the 192MB L2), f32 accumulate + f32 cell state.
// This revision: 2 N-tiles per wave sharing one A-fragment (halves A traffic,
// two independent WMMA chains for ILP) + global_prefetch on weight streams.
// ---------------------------------------------------------------------------

typedef __attribute__((ext_vector_type(16))) __bf16 v16bf;
typedef __attribute__((ext_vector_type(8)))  float  v8f;

#define BF16 __hip_bfloat16

union FragU { v16bf v; uint4 q[2]; };

// A fragment: row-major A[16][K], CDNA5 16-bit A 16x32 layout:
// lanes 0-15 (M=lane): K={0..7,16..23}; lanes 16-31 (M=lane&15): K={8..15,24..31}.
__device__ __forceinline__ v16bf load_frag_a(const BF16* p, int lda) {
  const int lane = threadIdx.x & 31;
  const int m    = lane & 15;
  const int koff = (lane >> 4) << 3;          // 0 or 8
  const unsigned short* row =
      reinterpret_cast<const unsigned short*>(p) + (size_t)m * (size_t)lda;
  FragU f;
  f.q[0] = *reinterpret_cast<const uint4*>(row + koff);
  f.q[1] = *reinterpret_cast<const uint4*>(row + 16 + koff);
  return f.v;
}

// B fragment from row-major weight W[N][K] (B = W^T): lanes 0-15 hold
// K={0..7,16..23} of column n=lane, lanes 16-31 hold K={8..15,24..31}.
__device__ __forceinline__ v16bf load_frag_b(const BF16* p, int ldw) {
  const int lane = threadIdx.x & 31;
  const int n    = lane & 15;
  const int koff = (lane >> 4) << 3;          // 0 or 8
  const unsigned short* row =
      reinterpret_cast<const unsigned short*>(p) + (size_t)n * (size_t)ldw + koff;
  FragU f;
  f.q[0] = *reinterpret_cast<const uint4*>(row);
  f.q[1] = *reinterpret_cast<const uint4*>(row + 16);
  return f.v;
}

__device__ __forceinline__ v8f wmma_acc(v16bf a, v16bf b, v8f c) {
  return __builtin_amdgcn_wmma_f32_16x16x32_bf16(false, a, false, b,
                                                 (short)0, c, false, false);
}

// Per-lane prefetch of this lane's weight row, `ahead` elements forward.
__device__ __forceinline__ void prefetch_b_row(const BF16* p, int ldw, int ahead) {
  const int lane = threadIdx.x & 31;
  __builtin_prefetch((const void*)(p + (size_t)(lane & 15) * (size_t)ldw + ahead),
                     0, 3);
}

__device__ __forceinline__ float sigm(float x) { return 1.0f / (1.0f + expf(-x)); }

#define V8F_ZERO {0.f,0.f,0.f,0.f,0.f,0.f,0.f,0.f}

// ---------------------------------------------------------------------------
// Generic GEMM: C[M,N] = act(A[M,K] @ W[N,K]^T + bias). Each wave owns a
// 16x32 output slab (two 16x16 N-tiles sharing one A-fragment); the second
// tile is skipped with a wave-uniform branch when N is not a multiple of 32.
// slot = blockIdx.y*wavesPerBlock + wave; n0 = slot*32.
// ---------------------------------------------------------------------------
__global__ void gemm_bf16_wmma(const BF16* __restrict__ A, int lda,
                               const BF16* __restrict__ W, int ldw,
                               const float* __restrict__ bias,
                               float* __restrict__ Cf, int ldcf,
                               BF16* __restrict__ Cb, int ldcb,
                               int K, int N, int relu) {
  const int wave  = threadIdx.x >> 5;
  const int lane  = threadIdx.x & 31;
  const int waves = blockDim.x >> 5;
  const int m0 = blockIdx.x * 16;
  const int n0 = (blockIdx.y * waves + wave) * 32;
  const bool has2 = (n0 + 16) < N;
  v8f acc0 = V8F_ZERO, acc1 = V8F_ZERO;
  const BF16* Ap  = A + (size_t)m0 * (size_t)lda;
  const BF16* Wp0 = W + (size_t)n0 * (size_t)ldw;
  const BF16* Wp1 = Wp0 + (size_t)16 * (size_t)ldw;
  if (has2) {
    for (int k0 = 0; k0 < K; k0 += 32) {
      const v16bf a = load_frag_a(Ap + k0, lda);
      acc0 = wmma_acc(a, load_frag_b(Wp0 + k0, ldw), acc0);
      acc1 = wmma_acc(a, load_frag_b(Wp1 + k0, ldw), acc1);
    }
  } else {
    for (int k0 = 0; k0 < K; k0 += 32)
      acc0 = wmma_acc(load_frag_a(Ap + k0, lda), load_frag_b(Wp0 + k0, ldw), acc0);
  }
  const int n  = lane & 15;
  const int hi = lane >> 4;
  const float bv0 = bias ? bias[n0 + n] : 0.f;
  const float bv1 = (bias && has2) ? bias[n0 + 16 + n] : 0.f;
#pragma unroll
  for (int r = 0; r < 8; ++r) {
    const int m = m0 + r + 8 * hi;
    float v0 = acc0[r] + bv0;
    if (relu) v0 = fmaxf(v0, 0.f);
    if (Cf) Cf[(size_t)m * (size_t)ldcf + n0 + n] = v0;
    if (Cb) Cb[(size_t)m * (size_t)ldcb + n0 + n] = __float2bfloat16(v0);
    if (has2) {
      float v1 = acc1[r] + bv1;
      if (relu) v1 = fmaxf(v1, 0.f);
      if (Cf) Cf[(size_t)m * (size_t)ldcf + n0 + 16 + n] = v1;
      if (Cb) Cb[(size_t)m * (size_t)ldcb + n0 + 16 + n] = __float2bfloat16(v1);
    }
  }
}

// ---------------------------------------------------------------------------
// LSTM step: gates = [A0|Ctx] @ Wih^T + H @ Whh^T + bih + bhh, fused i/f/g/o
// activation through LDS. grid = D/64 = 16 blocks, 256 threads.
// Wave w: mtile = w&1; processes 4 (gate,ntile) combos as 2 pairs; each pair
// shares A-fragments across two adjacent 16-col tiles (two WMMA chains).
// ---------------------------------------------------------------------------
__global__ void lstm_step_kernel(const BF16* __restrict__ A0, int lda0, int K0,
                                 const BF16* __restrict__ Ctx,         // [32][512]
                                 const BF16* __restrict__ Wih, int ldwih,
                                 const BF16* __restrict__ Hprev,       // [32][1024]
                                 const BF16* __restrict__ Whh,         // [4096][1024]
                                 const float* __restrict__ bih,
                                 const float* __restrict__ bhh,
                                 const float* __restrict__ Cprev,
                                 float* __restrict__ Cnext,
                                 BF16* __restrict__ Hnext,
                                 BF16* __restrict__ Hnext2, int ldh2) {
  __shared__ float gsm[4][32][65];
  const int wave = threadIdx.x >> 5;
  const int lane = threadIdx.x & 31;
  const int mt = wave & 1;
  const int m0 = mt * 16;
  const int cbase = (wave >> 1) * 4;   // 4 of 16 (gate,nt) combos per wave
  const BF16* Ap = A0 + (size_t)m0 * (size_t)lda0;
  const BF16* Cp = Ctx + (size_t)m0 * 512;
  const BF16* Hp = Hprev + (size_t)m0 * 1024;
  for (int i = 0; i < 2; ++i) {
    const int ca = cbase + i * 2;           // even combo
    const int g = ca >> 2, nt = ca & 3;     // pair shares gate g, nt and nt+1
    const int wrow0 = g * 1024 + blockIdx.x * 64 + nt * 16;
    const int wrow1 = wrow0 + 16;
    v8f acc0 = V8F_ZERO, acc1 = V8F_ZERO;
    {
      const BF16* W0 = Wih + (size_t)wrow0 * (size_t)ldwih;
      const BF16* W1 = Wih + (size_t)wrow1 * (size_t)ldwih;
      for (int k0 = 0; k0 < K0; k0 += 32) {
        prefetch_b_row(W0 + k0, ldwih, 128);
        const v16bf a = load_frag_a(Ap + k0, lda0);
        acc0 = wmma_acc(a, load_frag_b(W0 + k0, ldwih), acc0);
        acc1 = wmma_acc(a, load_frag_b(W1 + k0, ldwih), acc1);
      }
      const BF16* W0c = W0 + K0;
      const BF16* W1c = W1 + K0;
      for (int k0 = 0; k0 < 512; k0 += 32) {
        prefetch_b_row(W0c + k0, ldwih, 128);
        const v16bf a = load_frag_a(Cp + k0, 512);
        acc0 = wmma_acc(a, load_frag_b(W0c + k0, ldwih), acc0);
        acc1 = wmma_acc(a, load_frag_b(W1c + k0, ldwih), acc1);
      }
    }
    {
      const BF16* W0 = Whh + (size_t)wrow0 * 1024;
      const BF16* W1 = Whh + (size_t)wrow1 * 1024;
      for (int k0 = 0; k0 < 1024; k0 += 32) {
        prefetch_b_row(W0 + k0, 1024, 128);
        const v16bf a = load_frag_a(Hp + k0, 1024);
        acc0 = wmma_acc(a, load_frag_b(W0 + k0, 1024), acc0);
        acc1 = wmma_acc(a, load_frag_b(W1 + k0, 1024), acc1);
      }
    }
    const int n  = lane & 15;
    const int hi = lane >> 4;
    const float b0 = bih[wrow0 + n] + bhh[wrow0 + n];
    const float b1 = bih[wrow1 + n] + bhh[wrow1 + n];
#pragma unroll
    for (int r = 0; r < 8; ++r) {
      const int mrow = m0 + r + 8 * hi;
      gsm[g][mrow][nt * 16 + n]      = acc0[r] + b0;
      gsm[g][mrow][nt * 16 + 16 + n] = acc1[r] + b1;
    }
  }
  __syncthreads();
  const int col0 = blockIdx.x * 64;
  for (int e = threadIdx.x; e < 2048; e += blockDim.x) {
    const int m = e >> 6, c = e & 63;
    const float gi = gsm[0][m][c], gf = gsm[1][m][c];
    const float gg = gsm[2][m][c], go = gsm[3][m][c];
    const int gc = col0 + c;
    const float cn = sigm(gf) * Cprev[m * 1024 + gc] + sigm(gi) * tanhf(gg);
    const float hn = sigm(go) * tanhf(cn);
    Cnext[m * 1024 + gc] = cn;
    Hnext[m * 1024 + gc] = __float2bfloat16(hn);
    if (Hnext2) Hnext2[(size_t)m * (size_t)ldh2 + gc] = __float2bfloat16(hn);
  }
}

// ---------------------------------------------------------------------------
// Attention: location conv (K=31, 2ch, 32 filters) + energies + softmax +
// context. One block per batch element, 256 threads (thread = s).
// ---------------------------------------------------------------------------
__global__ void attn_step_kernel(const float* __restrict__ q,          // [32][128]
                                 const float* __restrict__ enc_cache,  // [32][256][128]
                                 const float* __restrict__ enc,        // [32][256][512]
                                 const unsigned char* __restrict__ mask,
                                 const float* __restrict__ W_loc,      // [32][2][31]
                                 const float* __restrict__ W_lp,       // [128][32]
                                 const float* __restrict__ v_en,       // [128]
                                 float* __restrict__ aw, float* __restrict__ cum,
                                 BF16* __restrict__ ctx_b,             // [32][512]
                                 BF16* __restrict__ npi,               // [32][1536]
                                 float* __restrict__ attn_out) {       // +b*800*256+s
  __shared__ float aw_sh[256], cum_sh[256], q_sh[128], v_sh[128], red[256];
  __shared__ float wloc_sh[1984];
  __shared__ float wlp_sh[4096];
  const int b = blockIdx.x, s = threadIdx.x;
  aw_sh[s]  = aw[b * 256 + s];
  cum_sh[s] = cum[b * 256 + s];
  if (s < 128) { q_sh[s] = q[b * 128 + s]; v_sh[s] = v_en[s]; }
  for (int i = s; i < 1984; i += 256) wloc_sh[i] = W_loc[i];
  for (int i = s; i < 4096; i += 256) wlp_sh[i]  = W_lp[i];
  __syncthreads();

  float feat[32];
#pragma unroll
  for (int nf = 0; nf < 32; ++nf) feat[nf] = 0.f;
  for (int j = 0; j < 31; ++j) {
    const int sp = s + j - 15;
    const bool in = (sp >= 0 && sp < 256);
    const float a0 = in ? aw_sh[sp]  : 0.f;
    const float a1 = in ? cum_sh[sp] : 0.f;
#pragma unroll
    for (int nf = 0; nf < 32; ++nf)
      feat[nf] += a0 * wloc_sh[nf * 62 + j] + a1 * wloc_sh[nf * 62 + 31 + j];
  }
  const float* cc = enc_cache + ((size_t)b * 256 + s) * 128;
  float e = 0.f;
  for (int a = 0; a < 128; ++a) {
    float loc = 0.f;
#pragma unroll
    for (int nf = 0; nf < 32; ++nf) loc += feat[nf] * wlp_sh[a * 32 + nf];
    e += tanhf(q_sh[a] + cc[a] + loc) * v_sh[a];
  }
  if (mask[b * 256 + s]) e = -1e9f;

  red[s] = e; __syncthreads();
  for (int off = 128; off > 0; off >>= 1) {
    if (s < off) red[s] = fmaxf(red[s], red[s + off]);
    __syncthreads();
  }
  const float mx = red[0]; __syncthreads();
  const float p = expf(e - mx);
  red[s] = p; __syncthreads();
  for (int off = 128; off > 0; off >>= 1) {
    if (s < off) red[s] += red[s + off];
    __syncthreads();
  }
  const float anew = p / red[0];
  const float cnew = cum_sh[s] + anew;
  __syncthreads();
  aw_sh[s] = anew;
  aw[b * 256 + s]  = anew;
  cum[b * 256 + s] = cnew;
  attn_out[(size_t)b * 800 * 256 + s] = anew;
  __syncthreads();

  for (int e0 = s; e0 < 512; e0 += 256) {
    float acc2 = 0.f;
    const float* ep = enc + (size_t)b * 256 * 512 + e0;
    for (int ss = 0; ss < 256; ++ss) acc2 += aw_sh[ss] * ep[(size_t)ss * 512];
    const BF16 hv = __float2bfloat16(acc2);
    ctx_b[b * 512 + e0] = hv;
    npi[(size_t)b * 1536 + 1024 + e0] = hv;
  }
}

__global__ void stop_kernel(const BF16* __restrict__ npi,
                            const float* __restrict__ Wstop,
                            const float* __restrict__ bstop,
                            float* __restrict__ out, int strideB) {
  const int b = threadIdx.x;
  const BF16* p = npi + (size_t)b * 1536;
  float s = bstop[0];
  for (int k = 0; k < 1536; ++k) s += __bfloat162float(p[k]) * Wstop[k];
  out[(size_t)b * strideB] = s;
}

// ---------------------------------------------------------------------------
// Postnet conv layer as 5 accumulated WMMA GEMMs over padded [B][T+4][Cip]
// bf16 activations; two Co tiles per wave share the X fragment; fused BN
// (eval) + optional tanh; last layer adds residual into f32 output.
// ---------------------------------------------------------------------------
__global__ void postnet_conv_kernel(const BF16* __restrict__ X, int Cip,
                                    const BF16* __restrict__ Wt,  // [5][Co][Cip]
                                    const float* __restrict__ bias,
                                    const float* __restrict__ gamma,
                                    const float* __restrict__ beta,
                                    const float* __restrict__ mean,
                                    const float* __restrict__ var,
                                    BF16* __restrict__ Y, int Cop,
                                    const float* __restrict__ melin,
                                    float* __restrict__ melout,
                                    int Co, int Tn, int dotanh) {
  const int wave  = threadIdx.x >> 5;
  const int lane  = threadIdx.x & 31;
  const int waves = blockDim.x >> 5;
  const int t0  = blockIdx.x * 16;
  const int co0 = (blockIdx.y * waves + wave) * 32;
  const int b   = blockIdx.z;
  const bool has2 = (co0 + 16) < Co;
  v8f acc0 = V8F_ZERO, acc1 = V8F_ZERO;
  const BF16* Xb = X + ((size_t)b * (Tn + 4) + t0) * (size_t)Cip;
  for (int k = 0; k < 5; ++k) {
    const BF16* Ak = Xb + (size_t)k * Cip;
    const BF16* W0 = Wt + ((size_t)k * Co + co0) * (size_t)Cip;
    const BF16* W1 = W0 + (size_t)16 * (size_t)Cip;
    if (has2) {
      for (int kc = 0; kc < Cip; kc += 32) {
        const v16bf a = load_frag_a(Ak + kc, Cip);
        acc0 = wmma_acc(a, load_frag_b(W0 + kc, Cip), acc0);
        acc1 = wmma_acc(a, load_frag_b(W1 + kc, Cip), acc1);
      }
    } else {
      for (int kc = 0; kc < Cip; kc += 32)
        acc0 = wmma_acc(load_frag_a(Ak + kc, Cip), load_frag_b(W0 + kc, Cip), acc0);
    }
  }
  const int n  = lane & 15;
  const int hi = lane >> 4;
#pragma unroll
  for (int p2 = 0; p2 < 2; ++p2) {
    if (p2 && !has2) break;                    // wave-uniform
    const v8f& acc = p2 ? acc1 : acc0;
    const int co = co0 + p2 * 16 + n;
    const float sc = gamma[co] * rsqrtf(var[co] + 1e-5f);
    const float sh = beta[co] - mean[co] * sc;
    const float bv = bias[co];
#pragma unroll
    for (int r = 0; r < 8; ++r) {
      const int t = t0 + r + 8 * hi;
      float y = (acc[r] + bv) * sc + sh;
      if (dotanh) y = tanhf(y);
      if (Y) {
        Y[((size_t)b * (Tn + 4) + t + 2) * (size_t)Cop + co] = __float2bfloat16(y);
      } else {
        const size_t o = ((size_t)b * Tn + t) * 80 + co;
        melout[o] = melin[o] + y;
      }
    }
  }
}

// ---------------------------------------------------------------------------
// Small setup kernels
// ---------------------------------------------------------------------------
__global__ void convert_pad_kernel(const float* __restrict__ src,
                                   BF16* __restrict__ dst,
                                   int rows, int Ks, int Kd) {
  const long i = (long)blockIdx.x * blockDim.x + threadIdx.x;
  if (i >= (long)rows * Kd) return;
  const int r = (int)(i / Kd), k = (int)(i % Kd);
  dst[i] = __float2bfloat16(k < Ks ? src[(long)r * Ks + k] : 0.f);
}

__global__ void build_prenet_input(const float* __restrict__ mels,
                                   BF16* __restrict__ dst) {  // [32][800][96]
  const long i = (long)blockIdx.x * blockDim.x + threadIdx.x;
  if (i >= 32L * 800 * 96) return;
  const int k = (int)(i % 96);
  const long rt = i / 96;
  const int t = (int)(rt % 800), b = (int)(rt / 800);
  const float v = (k < 80 && t > 0) ? mels[((long)b * 800 + (t - 1)) * 80 + k] : 0.f;
  dst[i] = __float2bfloat16(v);
}

__global__ void transpose_pnw_kernel(const float* __restrict__ src, // [Co][Ci][5]
                                     BF16* __restrict__ dst,        // [5][Co][Cip]
                                     int Co, int Ci, int Cip) {
  const long i = (long)blockIdx.x * blockDim.x + threadIdx.x;
  if (i >= 5L * Co * Cip) return;
  const int ci = (int)(i % Cip);
  const long r = i / Cip;
  const int co = (int)(r % Co), k = (int)(r / Co);
  const float v = (ci < Ci) ? src[((long)co * Ci + ci) * 5 + k] : 0.f;
  dst[i] = __float2bfloat16(v);
}

__global__ void zero_f32_kernel(float* p, long n) {
  const long i = (long)blockIdx.x * blockDim.x + threadIdx.x;
  if (i < n) p[i] = 0.f;
}
__global__ void zero_bf16_kernel(BF16* p, long n) {
  const long i = (long)blockIdx.x * blockDim.x + threadIdx.x;
  if (i < n) p[i] = __float2bfloat16(0.f);
}

// ---------------------------------------------------------------------------
// Host driver
// ---------------------------------------------------------------------------
extern "C" void kernel_launch(void* const* d_in, const int* in_sizes, int n_in,
                              void* d_out, int out_size, void* d_ws, size_t ws_size,
                              hipStream_t stream) {
  const float* enc    = (const float*)d_in[0];
  const unsigned char* emask = (const unsigned char*)d_in[1];
  const float* mels   = (const float*)d_in[2];
  const float* W_pre1 = (const float*)d_in[4];
  const float* W_pre2 = (const float*)d_in[5];
  const float* Wih0   = (const float*)d_in[6];
  const float* Whh0   = (const float*)d_in[7];
  const float* bih0   = (const float*)d_in[8];
  const float* bhh0   = (const float*)d_in[9];
  const float* Wih1   = (const float*)d_in[10];
  const float* Whh1   = (const float*)d_in[11];
  const float* bih1   = (const float*)d_in[12];
  const float* bhh1   = (const float*)d_in[13];
  const float* W_in   = (const float*)d_in[14];
  const float* b_in   = (const float*)d_in[15];
  const float* W_encp = (const float*)d_in[16];
  const float* W_loc  = (const float*)d_in[17];
  const float* W_lp   = (const float*)d_in[18];
  const float* v_en   = (const float*)d_in[19];
  const float* W_mel  = (const float*)d_in[20];
  const float* b_mel  = (const float*)d_in[21];
  const float* W_stop = (const float*)d_in[22];
  const float* b_stop = (const float*)d_in[23];
  const float *pnW[5], *pnB[5], *pnG[5], *pnBe[5], *pnM[5], *pnV[5];
  for (int i = 0; i < 5; ++i) {
    pnW[i]  = (const float*)d_in[24 + i];
    pnB[i]  = (const float*)d_in[29 + i];
    pnG[i]  = (const float*)d_in[34 + i];
    pnBe[i] = (const float*)d_in[39 + i];
    pnM[i]  = (const float*)d_in[44 + i];
    pnV[i]  = (const float*)d_in[49 + i];
  }

  float* d_mel  = (float*)d_out;              // [32][800][80]
  float* d_post = d_mel + 32 * 800 * 80;      // [32][800][80]
  float* d_stop = d_post + 32 * 800 * 80;     // [32][800]
  float* d_attn = d_stop + 32 * 800;          // [32][800][256]

  char* base = (char*)d_ws;
  size_t cur = 0;
  auto alloc = [&](size_t bytes) -> char* {
    char* p = base + cur;
    cur += (bytes + 255) & ~(size_t)255;
    return p;
  };
  BF16* Wpre1b = (BF16*)alloc(256 * 96 * 2);
  BF16* Wpre2b = (BF16*)alloc(256 * 256 * 2);
  BF16* Wih0b  = (BF16*)alloc((size_t)4096 * 768 * 2);
  BF16* Whh0b  = (BF16*)alloc((size_t)4096 * 1024 * 2);
  BF16* Wih1b  = (BF16*)alloc((size_t)4096 * 1536 * 2);
  BF16* Whh1b  = (BF16*)alloc((size_t)4096 * 1024 * 2);
  BF16* Winb   = (BF16*)alloc(128 * 1024 * 2);
  BF16* Wencb  = (BF16*)alloc(128 * 512 * 2);
  BF16* Wmelb  = (BF16*)alloc(80 * 1536 * 2);
  BF16* encb   = (BF16*)alloc((size_t)8192 * 512 * 2);
  float* enccf = (float*)alloc((size_t)8192 * 128 * 4);
  BF16* X0b    = (BF16*)alloc((size_t)25600 * 96 * 2);
  BF16* pre1b  = (BF16*)alloc((size_t)25600 * 256 * 2);
  BF16* xseqb  = (BF16*)alloc((size_t)25600 * 256 * 2);
  BF16*  h0b[2]; float* c0f[2]; BF16* h1b[2]; float* c1f[2];
  for (int i = 0; i < 2; ++i) {
    h0b[i] = (BF16*)alloc(32 * 1024 * 2);
    c0f[i] = (float*)alloc(32 * 1024 * 4);
    h1b[i] = (BF16*)alloc(32 * 1024 * 2);
    c1f[i] = (float*)alloc(32 * 1024 * 4);
  }
  BF16*  ctxb = (BF16*)alloc(32 * 512 * 2);
  float* awf  = (float*)alloc(32 * 256 * 4);
  float* cumf = (float*)alloc(32 * 256 * 4);
  float* qf   = (float*)alloc(32 * 128 * 4);
  BF16*  npib = (BF16*)alloc(32 * 1536 * 2);
  BF16* P0 = (BF16*)alloc((size_t)32 * 804 * 96 * 2);
  BF16* P1 = (BF16*)alloc((size_t)32 * 804 * 512 * 2);
  BF16* P2 = (BF16*)alloc((size_t)32 * 804 * 512 * 2);
  BF16* Wt0 = (BF16*)alloc((size_t)5 * 512 * 96 * 2);
  BF16* Wt1 = (BF16*)alloc((size_t)5 * 512 * 512 * 2);
  BF16* Wt2 = (BF16*)alloc((size_t)5 * 512 * 512 * 2);
  BF16* Wt3 = (BF16*)alloc((size_t)5 * 512 * 512 * 2);
  BF16* Wt4 = (BF16*)alloc((size_t)5 * 80 * 512 * 2);
  (void)ws_size; (void)in_sizes; (void)n_in; (void)out_size;

  auto cdiv = [](long a, long b) { return (int)((a + b - 1) / b); };

  // ---- weight conversions (deterministic every call) ----
  convert_pad_kernel<<<cdiv(256L * 96, 256), 256, 0, stream>>>(W_pre1, Wpre1b, 256, 80, 96);
  convert_pad_kernel<<<cdiv(256L * 256, 256), 256, 0, stream>>>(W_pre2, Wpre2b, 256, 256, 256);
  convert_pad_kernel<<<cdiv(4096L * 768, 256), 256, 0, stream>>>(Wih0, Wih0b, 4096, 768, 768);
  convert_pad_kernel<<<cdiv(4096L * 1024, 256), 256, 0, stream>>>(Whh0, Whh0b, 4096, 1024, 1024);
  convert_pad_kernel<<<cdiv(4096L * 1536, 256), 256, 0, stream>>>(Wih1, Wih1b, 4096, 1536, 1536);
  convert_pad_kernel<<<cdiv(4096L * 1024, 256), 256, 0, stream>>>(Whh1, Whh1b, 4096, 1024, 1024);
  convert_pad_kernel<<<cdiv(128L * 1024, 256), 256, 0, stream>>>(W_in, Winb, 128, 1024, 1024);
  convert_pad_kernel<<<cdiv(128L * 512, 256), 256, 0, stream>>>(W_encp, Wencb, 128, 512, 512);
  convert_pad_kernel<<<cdiv(80L * 1536, 256), 256, 0, stream>>>(W_mel, Wmelb, 80, 1536, 1536);
  convert_pad_kernel<<<cdiv(8192L * 512, 256), 256, 0, stream>>>(enc, encb, 8192, 512, 512);
  transpose_pnw_kernel<<<cdiv(5L * 512 * 96, 256), 256, 0, stream>>>(pnW[0], Wt0, 512, 80, 96);
  transpose_pnw_kernel<<<cdiv(5L * 512 * 512, 256), 256, 0, stream>>>(pnW[1], Wt1, 512, 512, 512);
  transpose_pnw_kernel<<<cdiv(5L * 512 * 512, 256), 256, 0, stream>>>(pnW[2], Wt2, 512, 512, 512);
  transpose_pnw_kernel<<<cdiv(5L * 512 * 512, 256), 256, 0, stream>>>(pnW[3], Wt3, 512, 512, 512);
  transpose_pnw_kernel<<<cdiv(5L * 80 * 512, 256), 256, 0, stream>>>(pnW[4], Wt4, 80, 512, 512);

  // ---- zero initial state + postnet pads ----
  zero_bf16_kernel<<<cdiv(32L * 1024, 256), 256, 0, stream>>>(h0b[0], 32L * 1024);
  zero_bf16_kernel<<<cdiv(32L * 1024, 256), 256, 0, stream>>>(h1b[0], 32L * 1024);
  zero_f32_kernel<<<cdiv(32L * 1024, 256), 256, 0, stream>>>(c0f[0], 32L * 1024);
  zero_f32_kernel<<<cdiv(32L * 1024, 256), 256, 0, stream>>>(c1f[0], 32L * 1024);
  zero_bf16_kernel<<<cdiv(32L * 512, 256), 256, 0, stream>>>(ctxb, 32L * 512);
  zero_f32_kernel<<<cdiv(32L * 256, 256), 256, 0, stream>>>(awf, 32L * 256);
  zero_f32_kernel<<<cdiv(32L * 256, 256), 256, 0, stream>>>(cumf, 32L * 256);
  zero_bf16_kernel<<<cdiv(32L * 804 * 96, 256), 256, 0, stream>>>(P0, 32L * 804 * 96);
  zero_bf16_kernel<<<cdiv(32L * 804 * 512, 256), 256, 0, stream>>>(P1, 32L * 804 * 512);
  zero_bf16_kernel<<<cdiv(32L * 804 * 512, 256), 256, 0, stream>>>(P2, 32L * 804 * 512);

  // ---- prenet + encoder projection cache ----
  build_prenet_input<<<cdiv(32L * 800 * 96, 256), 256, 0, stream>>>(mels, X0b);
  gemm_bf16_wmma<<<dim3(1600, 2), 128, 0, stream>>>(X0b, 96, Wpre1b, 96, nullptr,
      nullptr, 0, pre1b, 256, 96, 256, 1);
  gemm_bf16_wmma<<<dim3(1600, 2), 128, 0, stream>>>(pre1b, 256, Wpre2b, 256, nullptr,
      nullptr, 0, xseqb, 256, 256, 256, 1);
  gemm_bf16_wmma<<<dim3(512, 1), 128, 0, stream>>>(encb, 512, Wencb, 512, nullptr,
      enccf, 128, nullptr, 0, 512, 128, 0);

  // ---- sequential decoder scan ----
  for (int t = 0; t < 800; ++t) {
    const int pv = t & 1, nx = pv ^ 1;
    lstm_step_kernel<<<16, 256, 0, stream>>>(xseqb + (size_t)t * 256, 800 * 256, 256,
        ctxb, Wih0b, 768, h0b[pv], Whh0b, bih0, bhh0, c0f[pv], c0f[nx], h0b[nx],
        nullptr, 0);
    gemm_bf16_wmma<<<dim3(2, 1), 128, 0, stream>>>(h0b[nx], 1024, Winb, 1024, b_in,
        qf, 128, nullptr, 0, 1024, 128, 0);
    attn_step_kernel<<<32, 256, 0, stream>>>(qf, enccf, enc, emask, W_loc, W_lp,
        v_en, awf, cumf, ctxb, npib, d_attn + (size_t)t * 256);
    lstm_step_kernel<<<16, 256, 0, stream>>>(h0b[nx], 1024, 1024, ctxb, Wih1b, 1536,
        h1b[pv], Whh1b, bih1, bhh1, c1f[pv], c1f[nx], h1b[nx], npib, 1536);
    gemm_bf16_wmma<<<dim3(2, 1), 96, 0, stream>>>(npib, 1536, Wmelb, 1536, b_mel,
        d_mel + (size_t)t * 80, 800 * 80, P0 + (size_t)(t + 2) * 96, 804 * 96,
        1536, 80, 0);
    stop_kernel<<<1, 32, 0, stream>>>(npib, W_stop, b_stop, d_stop + t, 800);
  }

  // ---- postnet (residual) ----
  postnet_conv_kernel<<<dim3(50, 4, 32), 128, 0, stream>>>(P0, 96, Wt0, pnB[0],
      pnG[0], pnBe[0], pnM[0], pnV[0], P1, 512, nullptr, nullptr, 512, 800, 1);
  postnet_conv_kernel<<<dim3(50, 4, 32), 128, 0, stream>>>(P1, 512, Wt1, pnB[1],
      pnG[1], pnBe[1], pnM[1], pnV[1], P2, 512, nullptr, nullptr, 512, 800, 1);
  postnet_conv_kernel<<<dim3(50, 4, 32), 128, 0, stream>>>(P2, 512, Wt2, pnB[2],
      pnG[2], pnBe[2], pnM[2], pnV[2], P1, 512, nullptr, nullptr, 512, 800, 1);
  postnet_conv_kernel<<<dim3(50, 4, 32), 128, 0, stream>>>(P1, 512, Wt3, pnB[3],
      pnG[3], pnBe[3], pnM[3], pnV[3], P2, 512, nullptr, nullptr, 512, 800, 1);
  postnet_conv_kernel<<<dim3(50, 1, 32), 96, 0, stream>>>(P2, 512, Wt4, pnB[4],
      pnG[4], pnBe[4], pnM[4], pnV[4], nullptr, 0, d_mel, d_post, 80, 800, 0);
}
